// TGN_11510512353380
// MI455X (gfx1250) — compile-verified, matrix operand order checked by reading, not run on hardware
//
#include <hip/hip_runtime.h>

#define S_LEN 16
#define BSZ   64
#define NN    10000
#define NPAD  10016
#define LAT   128
#define DH    257
#define DMSG  129
#define KMSG  160      // DMSG padded up to multiple of 32 for bf16 WMMA

typedef float  v8f   __attribute__((ext_vector_type(8)));
typedef __bf16 v16bf __attribute__((ext_vector_type(16)));

__device__ __forceinline__ float sigmoidf_(float v){ return 1.0f/(1.0f+__expf(-v)); }

// CDNA5 async global->LDS copy (ASYNCcnt-tracked), per ISA 15.18.3 / 08_async_tensor.
__device__ __forceinline__ void async_load_b128(unsigned lds_off, const void* gptr){
  asm volatile("global_load_async_to_lds_b128 %0, %1, off"
               :: "v"(lds_off), "v"((unsigned long long)(size_t)gptr) : "memory");
}
__device__ __forceinline__ void wait_async0(){
  asm volatile("s_wait_asynccnt 0x0" ::: "memory");
}

// ---------------- zero init of persistent memory + bf16 shadows ----------------
__global__ void k_zero(float* mem, __bf16* mem_bf, __bf16* mem_bf_T){
  int i = blockIdx.x*blockDim.x + threadIdx.x;
  int st = gridDim.x*blockDim.x;
  for (int j=i; j<NN*LAT;   j+=st) mem[j]      = 0.f;
  for (int j=i; j<NPAD*LAT; j+=st) mem_bf[j]   = (__bf16)0.f;
  for (int j=i; j<LAT*NPAD; j+=st) mem_bf_T[j] = (__bf16)0.f;
}

// -------- one-time: bf16 GRU weights, W_ih K-padded 129 -> 160 -----------------
__global__ void k_wprep(const float* __restrict__ W_ih, const float* __restrict__ W_hh,
                        __bf16* __restrict__ Wih_bf, __bf16* __restrict__ Whh_bf){
  int i = blockIdx.x*blockDim.x + threadIdx.x;
  int st = gridDim.x*blockDim.x;
  for (int j=i; j<384*KMSG; j+=st){
    int row=j/KMSG, k=j%KMSG;
    Wih_bf[j] = (__bf16)((k<DMSG) ? W_ih[row*DMSG+k] : 0.f);
  }
  for (int j=i; j<384*LAT; j+=st) Whh_bf[j] = (__bf16)W_hh[j];
}

// -------- build bf16 GRU input matrices MSG(128x160), H(128x128) ---------------
__global__ void k_gbuild(const float* __restrict__ x, const float* __restrict__ t,
                         const int* __restrict__ src, const int* __restrict__ tgt,
                         const float* __restrict__ time_w, const float* __restrict__ time_b,
                         const float* __restrict__ mem,
                         __bf16* __restrict__ MSG, __bf16* __restrict__ Hbf, int s)
{
  int bb = blockIdx.x, b = bb & 63;
  int idx = (bb < 64) ? src[s*BSZ + b] : tgt[s*BSZ + b];
  int l = threadIdx.x;
  float tv = t[((size_t)(s*BSZ + b))*NN + idx];
  MSG[bb*KMSG + 1 + l] = (__bf16)__cosf(tv*time_w[l] + time_b[l]);
  Hbf[bb*LAT + l]      = (__bf16)mem[idx*LAT + l];
  if (l==0)  MSG[bb*KMSG] = (__bf16)x[((size_t)(s*BSZ + b))*NN + idx];
  if (l<KMSG-DMSG) MSG[bb*KMSG + DMSG + l] = (__bf16)0.f;
}

// -------- GRU gate GEMMs via WMMA: GI=MSG@Wih^T, GH=H@Whh^T (128x384) ----------
__global__ void __launch_bounds__(128)
k_ggemm(const __bf16* __restrict__ MSG, const __bf16* __restrict__ Hbf,
        const __bf16* __restrict__ Wih_bf, const __bf16* __restrict__ Whh_bf,
        float* __restrict__ GI, float* __restrict__ GH)
{
  int tid=threadIdx.x, wv=tid>>5, lane=tid&31;
  int tile = blockIdx.x*4 + wv;        // 0..191 = 8 M-tiles x 24 N-tiles
  int mt = tile/24, nt = tile%24;
  int mrow = mt*16 + (lane&15);
  int nrow = nt*16 + (lane&15);
  int khalf = (lane>>4)*16;
  v8f accI = {}, accH = {};
  for (int kk=0; kk<KMSG; kk+=32){
    v16bf a = *(const v16bf*)(MSG    + (size_t)mrow*KMSG + kk + khalf);
    v16bf b = *(const v16bf*)(Wih_bf + (size_t)nrow*KMSG + kk + khalf);
    accI = __builtin_amdgcn_wmma_f32_16x16x32_bf16(false, a, false, b, (short)0, accI, false, false);
  }
  for (int kk=0; kk<LAT; kk+=32){
    v16bf a = *(const v16bf*)(Hbf    + (size_t)mrow*LAT + kk + khalf);
    v16bf b = *(const v16bf*)(Whh_bf + (size_t)nrow*LAT + kk + khalf);
    accH = __builtin_amdgcn_wmma_f32_16x16x32_bf16(false, a, false, b, (short)0, accH, false, false);
  }
  int lanehi = (lane>>4)*8;
  #pragma unroll
  for (int r=0;r<8;++r){
    int m = mt*16 + r + lanehi;
    int n = nt*16 + (lane&15);
    GI[m*384 + n] = accI[r];
    GH[m*384 + n] = accH[r];
  }
}

// -------- GRU gates (pointwise) ------------------------------------------------
__global__ void k_ggate(const int* __restrict__ src, const int* __restrict__ tgt,
                        const float* __restrict__ mem,
                        const float* __restrict__ GI, const float* __restrict__ GH,
                        const float* __restrict__ b_ih, const float* __restrict__ b_hh,
                        float* __restrict__ new_h, int s)
{
  int bb = blockIdx.x, b = bb & 63, l = threadIdx.x;
  int idx = (bb < 64) ? src[s*BSZ + b] : tgt[s*BSZ + b];
  float h = mem[idx*LAT + l];
  float gi0 = GI[bb*384 +       l] + b_ih[l];
  float gi1 = GI[bb*384 + 128 + l] + b_ih[128+l];
  float gi2 = GI[bb*384 + 256 + l] + b_ih[256+l];
  float gh0 = GH[bb*384 +       l] + b_hh[l];
  float gh1 = GH[bb*384 + 128 + l] + b_hh[128+l];
  float gh2 = GH[bb*384 + 256 + l] + b_hh[256+l];
  float r = sigmoidf_(gi0+gh0);
  float z = sigmoidf_(gi1+gh1);
  float n = tanhf(gi2 + r*gh2);
  new_h[bb*LAT + l] = (1.f - z)*n + z*h;
}

// -------- deterministic ordered scatter (src rows then tgt rows, last wins) ----
__global__ void k_scatter(const int* __restrict__ src, const int* __restrict__ tgt,
                          const float* __restrict__ new_h,
                          float* mem, __bf16* mem_bf, __bf16* mem_bf_T, int s)
{
  int l = threadIdx.x; // 128
  for (int b=0;b<BSZ;++b){
    int r = src[s*BSZ+b];
    float v = new_h[b*LAT + l];
    mem[r*LAT+l]=v; mem_bf[r*LAT+l]=(__bf16)v; mem_bf_T[(size_t)l*NPAD + r]=(__bf16)v;
  }
  for (int b=0;b<BSZ;++b){
    int r = tgt[s*BSZ+b];
    float v = new_h[(64+b)*LAT + l];
    mem[r*LAT+l]=v; mem_bf[r*LAT+l]=(__bf16)v; mem_bf_T[(size_t)l*NPAD + r]=(__bf16)v;
  }
}

// -------- q = target_h @ Wq + bq ; p = Wk @ q ; c0 = q.bk  (per batch row) -----
__global__ void k_qp(const float* __restrict__ x, const int* __restrict__ tgt,
                     const float* __restrict__ time_b, const float* __restrict__ mem,
                     const float* __restrict__ Wq, const float* __restrict__ bq,
                     const float* __restrict__ Wk, const float* __restrict__ bk,
                     float* px, __bf16* pm_bf, float* pd, float* c0, int s)
{
  __shared__ float th[DH];
  __shared__ float qv[LAT];
  __shared__ float red[LAT];
  int b = blockIdx.x, l = threadIdx.x;
  int r = tgt[s*BSZ + b];
  th[1+l]     = mem[r*LAT + l];
  th[1+LAT+l] = __cosf(time_b[l]);
  if (l==0) th[0] = x[((size_t)(s*BSZ+b))*NN + r];
  __syncthreads();
  float acc = bq[l];
  for (int d=0; d<DH; ++d) acc += th[d]*Wq[d*LAT + l];
  qv[l] = acc;
  __syncthreads();
  float p1=0.f, p2=0.f;
  for (int j=0;j<LAT;++j){
    float qj = qv[j];
    p1 += Wk[(l      )*LAT + j]*qj;   // d = l
    p2 += Wk[(l+LAT  )*LAT + j]*qj;   // d = l+128
  }
  if (l==0) px[b] = p1; else pm_bf[b*LAT + (l-1)] = (__bf16)p1;
  if (l==0) pm_bf[b*LAT + 127] = (__bf16)p2;      // d=128 -> pm[127]
  else      pd[b*LAT + (l-1)]  = p2;              // d=129..255 -> pd[0..126]
  if (l==0){
    float p3=0.f;
    for (int j=0;j<LAT;++j) p3 += Wk[256*LAT + j]*qv[j];
    pd[b*LAT + 127] = p3;                         // d=256 -> pd[127]
  }
  red[l] = qv[l]*bk[l];
  __syncthreads();
  for (int off=64; off>0; off>>=1){ if (l<off) red[l]+=red[l+off]; __syncthreads(); }
  if (l==0) c0[b] = red[0];
}

// -------- scores: WMMA (pm . mem) + fused VALU dt term + mask ------------------
__global__ void __launch_bounds__(128)
k_scores(const float* __restrict__ x, const float* __restrict__ t,
         const unsigned char* __restrict__ mask,
         const float* __restrict__ time_w, const float* __restrict__ time_b,
         const __bf16* __restrict__ pm_bf, const float* __restrict__ pd,
         const float* __restrict__ px, const float* __restrict__ c0,
         const __bf16* __restrict__ mem_bf, float* __restrict__ scores, int s)
{
  __shared__ float pd_s[BSZ*LAT];
  __shared__ float tw_s[LAT], tb_s[LAT];
  __shared__ float c0_s[BSZ], px_s[BSZ];
  int tid = threadIdx.x;
  // stage 32KB of pd via CDNA5 async global->LDS copies (256B per lane here)
  #pragma unroll
  for (int j=0;j<16;++j){
    int elt = tid*64 + j*4;
    async_load_b128((unsigned)(size_t)&pd_s[elt], pd + elt);
  }
  if (tid<LAT){ tw_s[tid]=time_w[tid]; tb_s[tid]=time_b[tid]; }
  if (tid<BSZ){ c0_s[tid]=c0[tid]; px_s[tid]=px[tid]; }
  wait_async0();
  __syncthreads();
  int wv = tid>>5, lane = tid&31;
  int nb = blockIdx.x*16;
  int mrow  = wv*16 + (lane & 15);
  int khalf = (lane>>4)*16;
  v8f acc = {};
  for (int kk=0; kk<LAT; kk+=32){
    v16bf a  = *(const v16bf*)(pm_bf  + (size_t)mrow*LAT + kk + khalf);
    v16bf bf = *(const v16bf*)(mem_bf + (size_t)(nb + (lane&15))*LAT + kk + khalf);
    acc = __builtin_amdgcn_wmma_f32_16x16x32_bf16(false, a, false, bf, (short)0, acc, false, false);
  }
  const float scale = 0.08838834764831845f; // 1/sqrt(128)
  int n = nb + (lane & 15);
  int lanehi = (lane>>4)*8;
  #pragma unroll
  for (int r=0;r<8;++r){
    int b = wv*16 + r + lanehi;            // C-matrix layout: M = r + 8*(lane/16)
    float tv  = t[((size_t)(s*BSZ + b))*NN + n];
    float sdt = 0.f;
    const float* pdb = pd_s + b*LAT;
    #pragma unroll 4
    for (int l=0;l<LAT;++l) sdt += pdb[l]*__cosf(tv*tw_s[l] + tb_s[l]);
    float xv = x[((size_t)(s*BSZ + b))*NN + n];
    float v  = (c0_s[b] + px_s[b]*xv + acc[r] + sdt)*scale;
    if (!mask[((size_t)(s*BSZ + b))*NN + n]) v = -1.0e9f;
    scores[(size_t)b*NN + n] = v;
  }
}

// -------- softmax (unnormalized exp + row sum + Sum e*x) -----------------------
__global__ void k_softmax(const float* __restrict__ x, float* __restrict__ scores,
                          __bf16* __restrict__ e_bf, float* row_sum, float* A1, int s)
{
  __shared__ float red[256];
  int b = blockIdx.x, tid = threadIdx.x;
  float m = -3.0e38f;
  for (int n=tid; n<NN; n+=256) m = fmaxf(m, scores[(size_t)b*NN+n]);
  red[tid]=m; __syncthreads();
  for (int off=128; off>0; off>>=1){ if(tid<off) red[tid]=fmaxf(red[tid],red[tid+off]); __syncthreads(); }
  m = red[0]; __syncthreads();
  float sum=0.f, sumx=0.f;
  for (int n=tid; n<NN; n+=256){
    float e = __expf(scores[(size_t)b*NN+n] - m);
    scores[(size_t)b*NN+n] = e;                 // buffer now holds e
    e_bf[(size_t)b*NPAD + n] = (__bf16)e;
    sum  += e;
    sumx += e * x[((size_t)(s*BSZ + b))*NN + n];
  }
  if (tid < (NPAD-NN)) e_bf[(size_t)b*NPAD + NN + tid] = (__bf16)0.f;
  red[tid]=sum; __syncthreads();
  for (int off=128; off>0; off>>=1){ if(tid<off) red[tid]+=red[tid+off]; __syncthreads(); }
  if (tid==0) row_sum[b]=red[0];
  __syncthreads();
  red[tid]=sumx; __syncthreads();
  for (int off=128; off>0; off>>=1){ if(tid<off) red[tid]+=red[tid+off]; __syncthreads(); }
  if (tid==0) A1[b]=red[0];
}

// -------- Amem = e @ mem : WMMA, K split over 8 chunks for occupancy -----------
__global__ void __launch_bounds__(32)
k_zmem(const __bf16* __restrict__ e_bf, const __bf16* __restrict__ mem_bf_T,
       float* __restrict__ Amem_part)
{
  int lane = threadIdx.x;
  int tile = blockIdx.x;        // 0..31
  int c    = blockIdx.y;        // 0..7 K-chunk
  int bt = tile >> 3;           // 0..3  (b tile)
  int lt = tile & 7;            // 0..7  (l tile)
  int arow  = bt*16 + (lane&15);
  int brow  = lt*16 + (lane&15);
  int khalf = (lane>>4)*16;
  int k0 = c*1280;
  int kend = (k0+1280 < NPAD) ? k0+1280 : NPAD;
  v8f acc = {};
  for (int kk=k0; kk<kend; kk+=32){
    __builtin_prefetch(e_bf + (size_t)arow*NPAD + kk + 256, 0, 1);
    v16bf a  = *(const v16bf*)(e_bf      + (size_t)arow*NPAD + kk + khalf);
    v16bf bf = *(const v16bf*)(mem_bf_T  + (size_t)brow*NPAD + kk + khalf);
    acc = __builtin_amdgcn_wmma_f32_16x16x32_bf16(false, a, false, bf, (short)0, acc, false, false);
  }
  int lanehi = (lane>>4)*8;
  #pragma unroll
  for (int r=0;r<8;++r){
    int b = bt*16 + r + lanehi;
    int l = lt*16 + (lane&15);
    Amem_part[((size_t)c*BSZ + b)*LAT + l] = acc[r];
  }
}

// -------- Adt partials: Sum_n e[b,n]*cos(t*w_l + b_l), LDS-staged --------------
__global__ void __launch_bounds__(128)
k_zdt(const float* __restrict__ t, const float* __restrict__ e,
      const float* __restrict__ time_w, const float* __restrict__ time_b,
      float* __restrict__ Adt_part, int s)
{
  __shared__ float st_[128], se_[128];
  int b = blockIdx.x, c = blockIdx.y, l = threadIdx.x;
  float w = time_w[l], tb = time_b[l];
  const float* tp = t + ((size_t)(s*BSZ+b))*NN;
  const float* ep = e + (size_t)b*NN;
  float acc = 0.f;
  int n0 = c*1250, n1 = n0+1250;
  for (int base=n0; base<n1; base+=128){
    int n = base + l;
    float tvv = 0.f, evv = 0.f;
    if (n < n1){ tvv = tp[n]; evv = ep[n]; }
    __syncthreads();
    st_[l] = tvv; se_[l] = evv;
    __syncthreads();
    #pragma unroll 4
    for (int k=0;k<128;++k) acc += se_[k]*__cosf(st_[k]*w + tb);  // padded e==0
  }
  Adt_part[((size_t)c*BSZ + b)*LAT + l] = acc;
}

// -------- finalize: z = bv + a1*Wv_x + am@Wv_m + ad@Wv_d ; logit ---------------
__global__ void k_final(const float* __restrict__ Amem_part, const float* __restrict__ Adt_part,
                        const float* __restrict__ row_sum, const float* __restrict__ A1,
                        const float* __restrict__ Wv, const float* __restrict__ bv,
                        const float* __restrict__ W_out, const float* __restrict__ b_out,
                        float* __restrict__ out, int s)
{
  __shared__ float am[LAT], ad[LAT], red[LAT];
  int b = blockIdx.x, l = threadIdx.x;
  float adv = 0.f, amv = 0.f;
  for (int c=0;c<8;++c){                         // fixed order -> deterministic
    adv += Adt_part [((size_t)c*BSZ+b)*LAT + l];
    amv += Amem_part[((size_t)c*BSZ+b)*LAT + l];
  }
  float inv = 1.0f/row_sum[b];
  am[l] = amv*inv;
  ad[l] = adv*inv;
  __syncthreads();
  float a1 = A1[b]*inv;
  float z = bv[l] + a1*Wv[l];           // Wv row 0 (x part)
  for (int d=0; d<LAT; ++d){
    z += am[d]*Wv[(1+d)*LAT + l];
    z += ad[d]*Wv[(129+d)*LAT + l];
  }
  red[l] = z * W_out[l];
  __syncthreads();
  for (int off=64; off>0; off>>=1){ if(l<off) red[l]+=red[l+off]; __syncthreads(); }
  if (l==0) out[s*BSZ + b] = red[0] + b_out[0];
}

extern "C" void kernel_launch(void* const* d_in, const int* in_sizes, int n_in,
                              void* d_out, int out_size, void* d_ws, size_t ws_size,
                              hipStream_t stream)
{
  (void)in_sizes; (void)n_in; (void)out_size; (void)ws_size;
  const float* x      = (const float*)d_in[0];
  const float* t      = (const float*)d_in[1];
  const int*   src    = (const int*)d_in[2];
  const int*   tgt    = (const int*)d_in[3];
  const unsigned char* mask = (const unsigned char*)d_in[4];
  const float* time_w = (const float*)d_in[5];
  const float* time_b = (const float*)d_in[6];
  const float* W_ih   = (const float*)d_in[7];
  const float* W_hh   = (const float*)d_in[8];
  const float* b_ih   = (const float*)d_in[9];
  const float* b_hh   = (const float*)d_in[10];
  const float* Wq     = (const float*)d_in[11];
  const float* bq     = (const float*)d_in[12];
  const float* Wk     = (const float*)d_in[13];
  const float* bk     = (const float*)d_in[14];
  const float* Wv     = (const float*)d_in[15];
  const float* bv     = (const float*)d_in[16];
  const float* W_out  = (const float*)d_in[17];
  const float* b_out  = (const float*)d_in[18];
  float* out = (float*)d_out;

  char* w = (char*)d_ws;
  size_t off = 0;
  auto alloc = [&](size_t bytes)->char* {
    off = (off + 255) & ~(size_t)255;
    char* p = w + off; off += bytes; return p;
  };
  float*  mem       = (float*) alloc((size_t)NN*LAT*4);
  __bf16* mem_bf    = (__bf16*)alloc((size_t)NPAD*LAT*2);
  __bf16* mem_bf_T  = (__bf16*)alloc((size_t)LAT*NPAD*2);
  __bf16* Wih_bf    = (__bf16*)alloc((size_t)384*KMSG*2);
  __bf16* Whh_bf    = (__bf16*)alloc((size_t)384*LAT*2);
  __bf16* MSGb      = (__bf16*)alloc((size_t)128*KMSG*2);
  __bf16* Hbf       = (__bf16*)alloc((size_t)128*LAT*2);
  float*  GI        = (float*) alloc((size_t)128*384*4);
  float*  GH        = (float*) alloc((size_t)128*384*4);
  float*  new_h     = (float*) alloc((size_t)128*LAT*4);
  float*  px        = (float*) alloc(BSZ*4);
  __bf16* pm_bf     = (__bf16*)alloc(BSZ*LAT*2);
  float*  pd        = (float*) alloc(BSZ*LAT*4);
  float*  c0        = (float*) alloc(BSZ*4);
  float*  scores    = (float*) alloc((size_t)BSZ*NN*4);
  __bf16* e_bf      = (__bf16*)alloc((size_t)BSZ*NPAD*2);
  float*  row_sum   = (float*) alloc(BSZ*4);
  float*  A1        = (float*) alloc(BSZ*4);
  float*  Amem_part = (float*) alloc((size_t)8*BSZ*LAT*4);
  float*  Adt_part  = (float*) alloc((size_t)8*BSZ*LAT*4);

  k_zero <<<256,256,0,stream>>>(mem, mem_bf, mem_bf_T);
  k_wprep<<<64, 256,0,stream>>>(W_ih, W_hh, Wih_bf, Whh_bf);
  for (int s=0; s<S_LEN; ++s){
    k_gbuild <<<128,128,0,stream>>>(x,t,src,tgt,time_w,time_b,mem,MSGb,Hbf,s);
    k_ggemm  <<<48, 128,0,stream>>>(MSGb,Hbf,Wih_bf,Whh_bf,GI,GH);
    k_ggate  <<<128,128,0,stream>>>(src,tgt,mem,GI,GH,b_ih,b_hh,new_h,s);
    k_scatter<<<1,  128,0,stream>>>(src,tgt,new_h,mem,mem_bf,mem_bf_T,s);
    k_qp     <<<64, 128,0,stream>>>(x,tgt,time_b,mem,Wq,bq,Wk,bk,px,pm_bf,pd,c0,s);
    k_scores <<<NN/16,128,0,stream>>>(x,t,mask,time_w,time_b,pm_bf,pd,px,c0,mem_bf,scores,s);
    k_softmax<<<64, 256,0,stream>>>(x,scores,e_bf,row_sum,A1,s);
    k_zmem   <<<dim3(32,8),32,0,stream>>>(e_bf,mem_bf_T,Amem_part);
    k_zdt    <<<dim3(64,8),128,0,stream>>>(t,scores,time_w,time_b,Adt_part,s);
    k_final  <<<64, 128,0,stream>>>(Amem_part,Adt_part,row_sum,A1,Wv,bv,W_out,b_out,out,s);
  }
}
